// MultiHeadSelfAttention_81295140978806
// MI455X (gfx1250) — matrix operand order repaired and checked by reading, not compiled, not run on hardware
//
#include <hip/hip_runtime.h>
#include <hip/hip_bf16.h>

typedef __bf16 bf16_t;
typedef __attribute__((ext_vector_type(4)))  __bf16 v4bf;
typedef __attribute__((ext_vector_type(8)))  __bf16 v8bf;
typedef __attribute__((ext_vector_type(16))) __bf16 v16bf;
typedef __attribute__((ext_vector_type(8)))  float  v8f;

#define SEQ    4096
#define DMODEL 1024
#define NHEAD  16
#define HDIM   64
#define NTOK   8192   // N * S  (N = 2)

static __device__ __forceinline__ v8f zero8() {
  v8f z;
#pragma unroll
  for (int i = 0; i < 8; ++i) z[i] = 0.0f;
  return z;
}

static __device__ __forceinline__ v8f wmma_bf16(v16bf a, v16bf b, v8f c) {
  // D = A(16x32 bf16) * B(32x16 bf16) + C(16x16 f32)
  return __builtin_amdgcn_wmma_f32_16x16x32_bf16(false, a, false, b, (short)0, c,
                                                 false, false);
}

// A fragment 16x32 bf16 (row-major source): lane m = lane&15, half = lane>>4.
// lane holds K = 16h..16h+15 (VGPRs 0-3: 16h..16h+7, VGPRs 4-7: 16h+8..16h+15).
static __device__ __forceinline__ v16bf load_A(const bf16_t* __restrict__ base,
                                               int row0, int ld, int k0, int lane) {
  int m = lane & 15, hf = lane >> 4;
  const bf16_t* p = base + (size_t)(row0 + m) * ld + k0;
  v8bf lo = *(const v8bf*)(p + hf * 16);
  v8bf hi = *(const v8bf*)(p + hf * 16 + 8);
  v16bf r;
#pragma unroll
  for (int i = 0; i < 8; ++i) { r[i] = lo[i]; r[8 + i] = hi[i]; }
  return r;
}

// B fragment 32x16 bf16: B[k][n] = base[(col0+n)*ld + k0 + k].
// lane n = lane&15, half = lane>>4; lane holds K = 16h..16h+15 (32 contiguous bytes).
static __device__ __forceinline__ v16bf load_B(const bf16_t* __restrict__ base,
                                               int col0, int ld, int k0, int lane) {
  int n = lane & 15, hf = lane >> 4;
  return *(const v16bf*)(base + (size_t)(col0 + n) * ld + k0 + hf * 16);
}

__global__ void __launch_bounds__(256) cvt_f32_to_bf16(const float* __restrict__ in,
                                                       bf16_t* __restrict__ out, int n4) {
  int i = blockIdx.x * 256 + threadIdx.x;
  if (i >= n4) return;
  float4 f = ((const float4*)in)[i];
  v4bf o;
  o[0] = (bf16_t)f.x; o[1] = (bf16_t)f.y; o[2] = (bf16_t)f.z; o[3] = (bf16_t)f.w;
  ((v4bf*)out)[i] = o;
}

// y = x @ W^T. Each wave: 16 tokens x 64 output cols (= one head slice).
// blockIdx.z selects Q/K/V. Q,K stored [n,h,s,hd]; V stored transposed [n,h,hd,s].
__global__ void __launch_bounds__(256) mhsa_qkv_proj(
    const bf16_t* __restrict__ xb,
    const bf16_t* __restrict__ wq, const bf16_t* __restrict__ wk,
    const bf16_t* __restrict__ wv,
    bf16_t* __restrict__ qb, bf16_t* __restrict__ kb, bf16_t* __restrict__ vtb) {
  int lane = threadIdx.x & 31;
  int wave = threadIdx.x >> 5;
  int m0 = blockIdx.x * 128 + wave * 16;   // token tile base (over N*S)
  int h  = blockIdx.y;                     // head -> output cols n0 = h*64
  int which = blockIdx.z;
  const bf16_t* W = (which == 0) ? wq : (which == 1) ? wk : wv;
  int n0 = h * HDIM;

  v8f acc[4];
#pragma unroll
  for (int t = 0; t < 4; ++t) acc[t] = zero8();

  for (int kk = 0; kk < DMODEL; kk += 32) {
    v16bf a = load_A(xb, m0, DMODEL, kk, lane);
#pragma unroll
    for (int t = 0; t < 4; ++t) {
      // W is [out,in] row-major, exactly the B-operand layout for x @ W^T.
      v16bf b = load_B(W, n0 + 16 * t, DMODEL, kk, lane);
      acc[t] = wmma_bf16(a, b, acc[t]);
    }
  }

  int col = lane & 15, hf = lane >> 4;
  int nI = m0 >> 12;                       // batch index (m0 / 4096)
  int sB = (m0 & (SEQ - 1)) + 8 * hf;      // sequence base for this half
  size_t headBase = (size_t)(nI * NHEAD + h);

  if (which < 2) {
    bf16_t* dst = ((which == 0) ? qb : kb) + headBase * SEQ * HDIM;
#pragma unroll
    for (int t = 0; t < 4; ++t) {
      int d = 16 * t + col;
#pragma unroll
      for (int v = 0; v < 8; ++v)
        dst[(size_t)(sB + v) * HDIM + d] = (bf16_t)acc[t][v];
    }
  } else {
    bf16_t* dst = vtb + headBase * HDIM * SEQ;  // [hd][s]
#pragma unroll
    for (int t = 0; t < 4; ++t) {
      int d = 16 * t + col;
      v8bf pk;
#pragma unroll
      for (int v = 0; v < 8; ++v) pk[v] = (bf16_t)acc[t][v];
      *(v8bf*)(dst + (size_t)d * SEQ + sB) = pk;   // 16B contiguous along s
    }
  }
}

// Flash attention: one wave = 16 queries of one (n,h), streaming keys in blocks of 32.
// scores^T = K·Q^T (lane owns a query column -> in-lane softmax), O^T = V^T·P^T.
__global__ void __launch_bounds__(256) mhsa_attention(
    const bf16_t* __restrict__ qb, const bf16_t* __restrict__ kb,
    const bf16_t* __restrict__ vtb, float* __restrict__ out) {
  int lane = threadIdx.x & 31;
  int wave = threadIdx.x >> 5;
  int gw = blockIdx.x * 8 + wave;
  int qt = gw & (SEQ / 16 - 1);            // query tile 0..255
  int nh = gw >> 8;                        // (n*H + h) 0..31
  int q0 = qt * 16;
  const bf16_t* q  = qb  + (size_t)nh * SEQ * HDIM;
  const bf16_t* k  = kb  + (size_t)nh * SEQ * HDIM;
  const bf16_t* vt = vtb + (size_t)nh * HDIM * SEQ;

  // Q^T B-fragments (loaded once): B[d][query]
  v16bf bq0 = load_B(q, q0, HDIM, 0,  lane);
  v16bf bq1 = load_B(q, q0, HDIM, 32, lane);

  v8f o[4];
#pragma unroll
  for (int j = 0; j < 4; ++j) o[j] = zero8();
  float mrow = -1e30f, lrow = 0.0f;
  const float sc = 0.125f;                 // 1/sqrt(hd)
  int hf = lane >> 4;

  for (int k0 = 0; k0 < SEQ; k0 += 32) {
    // scores^T: two 16key x 16query tiles, K-reduce over hd=64 (2 WMMAs each)
    v8f s0 = zero8(), s1 = zero8();
    s0 = wmma_bf16(load_A(k, k0,      HDIM, 0,  lane), bq0, s0);
    s0 = wmma_bf16(load_A(k, k0,      HDIM, 32, lane), bq1, s0);
    s1 = wmma_bf16(load_A(k, k0 + 16, HDIM, 0,  lane), bq0, s1);
    s1 = wmma_bf16(load_A(k, k0 + 16, HDIM, 32, lane), bq1, s1);

    // Online softmax. Lane owns query col (lane&15); its 16 score values cover
    // keys {v+8h} of each tile. Full row stats need one xor-16 shuffle.
    float t0[8], t1[8], bm = -1e30f;
#pragma unroll
    for (int i = 0; i < 8; ++i) {
      t0[i] = s0[i] * sc; t1[i] = s1[i] * sc;
      bm = fmaxf(bm, fmaxf(t0[i], t1[i]));
    }
    bm = fmaxf(bm, __shfl_xor(bm, 16, 32));
    float mnew = fmaxf(mrow, bm);
    float cf = __expf(mrow - mnew);
    float p0[8], p1[8], ps = 0.0f;
#pragma unroll
    for (int i = 0; i < 8; ++i) {
      p0[i] = __expf(t0[i] - mnew);
      p1[i] = __expf(t1[i] - mnew);
      ps += p0[i] + p1[i];
    }
    ps += __shfl_xor(ps, 16, 32);
    lrow = lrow * cf + ps;
    mrow = mnew;
#pragma unroll
    for (int j = 0; j < 4; ++j)
#pragma unroll
      for (int i = 0; i < 8; ++i) o[j][i] *= cf;

    // C-layout P^T -> B-operand (32key x 16query). Element e of lane (q, h)
    // is key 16h+e. Keys in the "other" half-source live in the partner lane
    // (xor 16); each direction of one shuffle carries the array the partner needs.
    v16bf pb;
#pragma unroll
    for (int i = 0; i < 8; ++i) {
      float send = hf ? p0[i] : p1[i];
      float recv = __shfl_xor(send, 16, 32);
      pb[i]     = (bf16_t)(hf ? recv : p0[i]);
      pb[8 + i] = (bf16_t)(hf ? p1[i] : recv);
    }

    // O^T += V^T(16d x 32key) @ P^T(32key x 16q), 4 d-tiles
#pragma unroll
    for (int j = 0; j < 4; ++j) {
      v16bf av = load_A(vt, 16 * j, SEQ, k0, lane);  // contiguous along keys
      o[j] = wmma_bf16(av, pb, o[j]);
    }
  }

  // Epilogue: scale by 1/l and write f32 out[(n, s=q, h*64+d)]
  float linv = 1.0f / lrow;
  int nI = nh >> 4, hh = nh & 15;
  int qq = q0 + (lane & 15);
  float* orow = out + ((size_t)nI * SEQ + qq) * DMODEL + hh * HDIM;
#pragma unroll
  for (int j = 0; j < 4; ++j) {
    v8f t;
#pragma unroll
    for (int v = 0; v < 8; ++v) t[v] = o[j][v] * linv;
    *(v8f*)(orow + 16 * j + 8 * hf) = t;   // 32B contiguous in d
  }
}

extern "C" void kernel_launch(void* const* d_in, const int* in_sizes, int n_in,
                              void* d_out, int out_size, void* d_ws, size_t ws_size,
                              hipStream_t stream) {
  (void)in_sizes; (void)n_in; (void)out_size; (void)ws_size;
  const float* x  = (const float*)d_in[0];
  const float* Wq = (const float*)d_in[1];
  const float* Wk = (const float*)d_in[2];
  const float* Wv = (const float*)d_in[3];
  float* out = (float*)d_out;

  // Workspace layout (bf16): xb | Wq | Wk | Wv | Q | K | V^T  (~73 MB)
  bf16_t* xb  = (bf16_t*)d_ws;
  bf16_t* wqb = xb  + (size_t)NTOK * DMODEL;
  bf16_t* wkb = wqb + (size_t)DMODEL * DMODEL;
  bf16_t* wvb = wkb + (size_t)DMODEL * DMODEL;
  bf16_t* qb  = wvb + (size_t)DMODEL * DMODEL;
  bf16_t* kb  = qb  + (size_t)NTOK * DMODEL;
  bf16_t* vtb = kb  + (size_t)NTOK * DMODEL;

  const int nx4 = NTOK * DMODEL / 4;        // 2,097,152
  const int nw4 = DMODEL * DMODEL / 4;      //   262,144
  cvt_f32_to_bf16<<<(nx4 + 255) / 256, 256, 0, stream>>>(x,  xb,  nx4);
  cvt_f32_to_bf16<<<(nw4 + 255) / 256, 256, 0, stream>>>(Wq, wqb, nw4);
  cvt_f32_to_bf16<<<(nw4 + 255) / 256, 256, 0, stream>>>(Wk, wkb, nw4);
  cvt_f32_to_bf16<<<(nw4 + 255) / 256, 256, 0, stream>>>(Wv, wvb, nw4);

  // 512 M-tiles x 16 heads x {Q,K,V}
  mhsa_qkv_proj<<<dim3(NTOK / 128, NHEAD, 3), 256, 0, stream>>>(
      xb, wqb, wkb, wvb, qb, kb, vtb);

  // 8192 waves = 2*16 heads * 256 query tiles, 8 waves/block
  mhsa_attention<<<dim3((2 * NHEAD * (SEQ / 16)) / 8), 256, 0, stream>>>(
      qb, kb, vtb, out);
}
